// GHCBlock_68143951118651
// MI455X (gfx1250) — matrix operand order repaired and checked by reading, not compiled, not run on hardware
//
#include <hip/hip_runtime.h>

// ---------------------------------------------------------------------------
// GNN hyper-conv block for MI455X (gfx1250, wave32, WMMA).
// Dense parts: H = gelu(X@W1^T+b1); T = gelu(H@WA^T); out = feat@W2^T+b2
//   -> v_wmma_f32_16x16x32_f16, f32 accumulate.
// Edge parts: per-node hyper weights G = T@WB^T gathered per edge; outer-
//   product scatter into agg(N,4,128) via f32 atomics (L2-resident, 102MB),
//   gelu, gather-contract, scatter-mean.
// ---------------------------------------------------------------------------

typedef __attribute__((ext_vector_type(16))) _Float16 v16h;
typedef __attribute__((ext_vector_type(8)))  _Float16 v8h;
typedef __attribute__((ext_vector_type(8)))  float    v8f;

#define NN   50000
#define NE   400000
#define HID  128
#define MIX  4

__device__ __forceinline__ float gelu_erf(float x) {
    return 0.5f * x * (1.0f + erff(x * 0.7071067811865476f));
}

__global__ void k_f32_to_f16(const float* __restrict__ in, _Float16* __restrict__ out, int n) {
    int i = blockIdx.x * blockDim.x + threadIdx.x;
    if (i < n) out[i] = (_Float16)in[i];
}

__global__ void k_zero_f32(float* __restrict__ p, int n) {
    int i = blockIdx.x * blockDim.x + threadIdx.x;
    if (i < n) p[i] = 0.0f;
}

// C(M x 128) = act(A(M x K) @ W(128 x K)^T + bias). M % 16 == 0, K % 32 == 0.
// 256 threads = 8 waves; wave w computes the 16x16 tile at columns [16w,16w+16).
// A frag (16x32 f16): lane m = lane&15; elems 0..7 = K k0+off..k0+off+7,
//   elems 8..15 = K k0+16+off.. (off = 0 for lanes<16, 8 for lanes>=16).
// B frag (32x16 f16): lane n = lane&15; same K layout as A frag.
__global__ void k_wmma_gemm(const _Float16* __restrict__ A,
                            const _Float16* __restrict__ Wt,
                            const float* __restrict__ bias,   // 128 or nullptr
                            float* __restrict__ Cf32,         // M x 128 or nullptr
                            _Float16* __restrict__ Cf16,      // M x 128 or nullptr
                            int K, int do_gelu) {
    const int lane = threadIdx.x & 31;
    const int wave = threadIdx.x >> 5;     // N-tile 0..7
    const int up   = lane >> 4;            // 0: lanes 0-15, 1: lanes 16-31
    const int l16  = lane & 15;

    const int arow = blockIdx.x * 16 + l16;
    const int bcol = wave * 16 + l16;

    const _Float16* pa = A  + (size_t)arow * K + (up ? 8 : 0);
    const _Float16* pb = Wt + (size_t)bcol * K + (up ? 8 : 0);

    v8f acc = {};
    for (int k0 = 0; k0 < K; k0 += 32) {
        union { v16h v; v8h h[2]; } af;
        af.h[0] = *(const v8h*)(pa + k0);        // 16B aligned
        af.h[1] = *(const v8h*)(pa + k0 + 16);
        union { v16h v; v8h h[2]; } bu;
        bu.h[0] = *(const v8h*)(pb + k0);        // 16B aligned
        bu.h[1] = *(const v8h*)(pb + k0 + 16);
        v16h bf = bu.v;
        acc = __builtin_amdgcn_wmma_f32_16x16x32_f16(
            /*neg_a=*/false, af.v, /*neg_b=*/false, bf,
            /*c_mod=*/(short)0, acc, /*reuse_a=*/false, /*reuse_b=*/false);
    }

    const float bv = bias ? bias[bcol] : 0.0f;
    const int mbase = blockIdx.x * 16 + (up ? 8 : 0);   // D: VGPR r -> M=r / r+8
    #pragma unroll
    for (int r = 0; r < 8; ++r) {
        float v = acc[r] + bv;
        if (do_gelu) v = gelu_erf(v);
        size_t idx = (size_t)(mbase + r) * HID + bcol;
        if (Cf32) Cf32[idx] = v;
        if (Cf16) Cf16[idx] = (_Float16)v;
    }
}

// G[n][m] = sum_k T[n][k] * WB[m][k]   (N x MIX, tiny GEMM -> scalar VALU)
__global__ void k_gnode(const _Float16* __restrict__ T, const float* __restrict__ WB,
                        float* __restrict__ G, int n_nodes) {
    int i = blockIdx.x * blockDim.x + threadIdx.x;   // n*MIX + m
    if (i >= n_nodes * MIX) return;
    int n = i >> 2, m = i & 3;
    const _Float16* t = T + (size_t)n * HID;
    const float* w = WB + m * HID;
    float s = 0.0f;
    #pragma unroll 4
    for (int k = 0; k < HID; ++k) s += (float)t[k] * w[k];
    G[i] = s;
}

// agg[dst][m][h] += H[src][h] * G[src][m];  cnt[dst] += 1. 128 threads/edge.
__global__ void k_edge_scatter(const int* __restrict__ src, const int* __restrict__ dst,
                               const float* __restrict__ H, const float* __restrict__ G,
                               float* __restrict__ agg, float* __restrict__ cnt, int E) {
    int e = blockIdx.x * 2 + (threadIdx.x >> 7);
    if (e >= E) return;
    int h = threadIdx.x & 127;
    int s = src[e], d = dst[e];
    float x = H[(size_t)s * HID + h];
    float* ab = agg + (size_t)d * (HID * MIX);
    #pragma unroll
    for (int m = 0; m < MIX; ++m)
        atomicAdd(ab + m * HID + h, x * G[s * MIX + m]);
    if (h == 0) atomicAdd(cnt + d, 1.0f);
}

__global__ void k_gelu_inplace(float* __restrict__ p, int n) {
    int i = blockIdx.x * blockDim.x + threadIdx.x;
    if (i < n) p[i] = gelu_erf(p[i]);
}

// agg_node[dst][h] += sum_m gelu_agg[dst][m][h] * G[src][m]
__global__ void k_edge_gather(const int* __restrict__ src, const int* __restrict__ dst,
                              const float* __restrict__ G, const float* __restrict__ agg,
                              float* __restrict__ agg_node, int E) {
    int e = blockIdx.x * 2 + (threadIdx.x >> 7);
    if (e >= E) return;
    int h = threadIdx.x & 127;
    int s = src[e], d = dst[e];
    const float* ab = agg + (size_t)d * (HID * MIX);
    float v = 0.0f;
    #pragma unroll
    for (int m = 0; m < MIX; ++m) v += ab[m * HID + h] * G[s * MIX + m];
    atomicAdd(agg_node + (size_t)d * HID + h, v);
}

// feat = gelu([agg_node/max(cnt,1), H]) as f16 (N x 256)
__global__ void k_finalize(const float* __restrict__ agg_node, const float* __restrict__ cnt,
                           const float* __restrict__ H, _Float16* __restrict__ feat) {
    int i = blockIdx.x * blockDim.x + threadIdx.x;
    if (i >= NN * HID) return;
    int n = i >> 7, h = i & 127;
    float c = fmaxf(cnt[n], 1.0f);
    feat[(size_t)n * 256 + h]       = (_Float16)gelu_erf(agg_node[i] / c);
    feat[(size_t)n * 256 + 128 + h] = (_Float16)gelu_erf(H[i]);
}

extern "C" void kernel_launch(void* const* d_in, const int* in_sizes, int n_in,
                              void* d_out, int out_size, void* d_ws, size_t ws_size,
                              hipStream_t stream) {
    const float* X  = (const float*)d_in[0];
    const int*   src = (const int*)d_in[1];          // edge_index (2, E)
    const int*   dst = src + NE;
    const float* W1 = (const float*)d_in[2];
    const float* b1 = (const float*)d_in[3];
    const float* WA = (const float*)d_in[4];
    const float* WB = (const float*)d_in[5];
    const float* W2 = (const float*)d_in[6];
    const float* b2 = (const float*)d_in[7];
    float* out = (float*)d_out;

    char* w = (char*)d_ws;
    size_t off = 0;
    auto alloc = [&](size_t bytes) -> void* {
        void* p = w + off;
        off += (bytes + 255) & ~(size_t)255;
        return p;
    };
    _Float16* Xh   = (_Float16*)alloc((size_t)NN * HID * 2);
    _Float16* W1h  = (_Float16*)alloc((size_t)HID * HID * 2);
    _Float16* WAh  = (_Float16*)alloc((size_t)HID * HID * 2);
    _Float16* W2h  = (_Float16*)alloc((size_t)HID * 2 * HID * 2);
    float*    Hf   = (float*)   alloc((size_t)NN * HID * 4);
    _Float16* Hh   = (_Float16*)alloc((size_t)NN * HID * 2);
    _Float16* Th   = (_Float16*)alloc((size_t)NN * HID * 2);
    float*    G    = (float*)   alloc((size_t)NN * MIX * 4);
    float*    agg  = (float*)   alloc((size_t)NN * HID * MIX * 4);
    float*    cnt  = (float*)   alloc((size_t)NN * 4);
    float*    aggn = (float*)   alloc((size_t)NN * HID * 4);
    _Float16* feath= (_Float16*)alloc((size_t)NN * 2 * HID * 2);

    // f32 -> f16 conversions
    k_f32_to_f16<<<(NN * HID + 255) / 256, 256, 0, stream>>>(X,  Xh,  NN * HID);
    k_f32_to_f16<<<(HID * HID + 255) / 256, 256, 0, stream>>>(W1, W1h, HID * HID);
    k_f32_to_f16<<<(HID * HID + 255) / 256, 256, 0, stream>>>(WA, WAh, HID * HID);
    k_f32_to_f16<<<(HID * 2 * HID + 255) / 256, 256, 0, stream>>>(W2, W2h, HID * 2 * HID);

    // zero accumulators (harness poisons ws; must re-zero every call)
    k_zero_f32<<<(NN * HID * MIX + 255) / 256, 256, 0, stream>>>(agg,  NN * HID * MIX);
    k_zero_f32<<<(NN + 255) / 256, 256, 0, stream>>>(cnt, NN);
    k_zero_f32<<<(NN * HID + 255) / 256, 256, 0, stream>>>(aggn, NN * HID);

    // H = gelu(X @ W1^T + b1), keep f32 + f16 copies
    k_wmma_gemm<<<NN / 16, 256, 0, stream>>>(Xh, W1h, b1, Hf, Hh, HID, 1);
    // T = gelu(H @ WA^T)  (f16 only, feeds tiny GEMM)
    k_wmma_gemm<<<NN / 16, 256, 0, stream>>>(Hh, WAh, nullptr, nullptr, Th, HID, 1);
    // G = T @ WB^T  (per-node hyper weights; gathered per edge)
    k_gnode<<<(NN * MIX + 255) / 256, 256, 0, stream>>>(Th, WB, G, NN);

    // scatter outer(H[src], G[src]) into agg(dst), count edges per dst
    k_edge_scatter<<<NE / 2, 256, 0, stream>>>(src, dst, Hf, G, agg, cnt, NE);
    // agg = gelu(agg)
    k_gelu_inplace<<<(NN * HID * MIX + 255) / 256, 256, 0, stream>>>(agg, NN * HID * MIX);
    // out_edge contract over mix, scatter-sum into agg_node
    k_edge_gather<<<NE / 2, 256, 0, stream>>>(src, dst, G, agg, aggn, NE);
    // scatter-mean + concat + gelu -> feat f16 (N x 256)
    k_finalize<<<(NN * HID + 255) / 256, 256, 0, stream>>>(aggn, cnt, Hf, feath);

    // out = feat @ W2^T + b2
    k_wmma_gemm<<<NN / 16, 256, 0, stream>>>(feath, W2h, b2, out, nullptr, 2 * HID, 0);
}
